// GeneratorBlock_17351667876208
// MI455X (gfx1250) — matrix operand order repaired and checked by reading, not compiled, not run on hardware
//
#include <hip/hip_runtime.h>
#include <math.h>
#include <stdint.h>

typedef __bf16 bf16_t;
typedef __attribute__((ext_vector_type(16))) __bf16 v16bf;
typedef __attribute__((ext_vector_type(8)))  __bf16 v8bf;
typedef __attribute__((ext_vector_type(8)))  float  v8f;
typedef __attribute__((ext_vector_type(4)))  int    v4i;

#define AS1 __attribute__((address_space(1)))
#define AS3 __attribute__((address_space(3)))

#if defined(__HIP_DEVICE_COMPILE__) && __has_builtin(__builtin_amdgcn_global_load_async_to_lds_b128)
#define HAVE_ASYNC_LDS 1
#else
#define HAVE_ASYNC_LDS 0
#endif
#if defined(__HIP_DEVICE_COMPILE__) && __has_builtin(__builtin_amdgcn_s_wait_asynccnt)
#define HAVE_WAIT_ASYNC 1
#else
#define HAVE_WAIT_ASYNC 0
#endif

namespace {

constexpr int NB  = 4;     // batch
constexpr int C   = 256;   // channels (half-state width, also conv in/out)
constexpr int HW  = 4096;  // 64*64 (all compute at 64-res; 2x2 upsample at the end)
constexpr int SD  = 512;   // style dim
constexpr int TM  = 128;   // workgroup M tile (hw)
constexpr int TN  = 128;   // workgroup N tile (out ch)
constexpr int KB  = 64;    // K block staged in LDS
constexpr int LROW = 72;   // padded LDS row stride in bf16 (72*2=144B -> conflict-free b128)

__device__ __forceinline__ float  bf2f(bf16_t x) { return (float)x; }
__device__ __forceinline__ bf16_t f2bf(float x)  { return (bf16_t)x; }

union Frag { uint4 u[2]; v16bf v; };

__device__ __forceinline__ void g2l_b128(const bf16_t* gp, bf16_t* lp) {
#if HAVE_ASYNC_LDS
  __builtin_amdgcn_global_load_async_to_lds_b128(
      (AS1 v4i*)(uintptr_t)gp, (AS3 v4i*)(uint32_t)(uintptr_t)lp, 0, 0);
#else
  *(uint4*)lp = *(const uint4*)gp;
#endif
}

__device__ __forceinline__ void lds_fill_wait() {
#if HAVE_ASYNC_LDS
#if HAVE_WAIT_ASYNC
  __builtin_amdgcn_s_wait_asynccnt(0);
#else
  asm volatile("s_wait_asynccnt 0" ::: "memory");
#endif
#endif
  __syncthreads();
}

// D[hw, o] = sum_c A[b, hw, c] * B[o, c]   (A channels-last bf16, B row-major bf16)
// MODE 0 (conv):  v += bias[o]; write channel-duplicated into X1/X2 (bf16).
// MODE 1 (layer): h = g[b,o]*v; X2[b,hw,o] += gelu_exact(h)  (in place).
template <int MODE>
__global__ __launch_bounds__(256) void gemm_wmma(
    const bf16_t* __restrict__ A, const bf16_t* __restrict__ B,
    const float* __restrict__ bias, const float* __restrict__ g,
    bf16_t* __restrict__ X1, bf16_t* __restrict__ X2)
{
  __shared__ __align__(16) bf16_t sA[TM * LROW];
  __shared__ __align__(16) bf16_t sB[TN * LROW];

  const int t    = threadIdx.x;
  const int lane = t & 31;
  const int wid  = t >> 5;
  const int half = lane >> 4;
  const int l15  = lane & 15;
  const int wm   = (wid & 3) * 32;   // 4 waves along M
  const int wn   = (wid >> 2) * 64;  // 2 waves along N
  const int b    = blockIdx.z;
  const int m0   = blockIdx.x * TM;
  const int n0   = blockIdx.y * TN;

  const bf16_t* Ab = A + ((size_t)b * HW + m0) * C;
  const bf16_t* Bb = B + (size_t)n0 * C;

  v8f acc[2][4] = {};

  for (int kb = 0; kb < C; kb += KB) {
    if (kb) __syncthreads();
    // 128 rows x 64 bf16 = 1024 16B chunks per matrix; 4 per thread
#pragma unroll
    for (int j = 0; j < 4; ++j) {
      int cc  = t + j * 256;
      int row = cc >> 3;
      int col = (cc & 7) * 8;
      g2l_b128(Ab + (size_t)row * C + kb + col, &sA[row * LROW + col]);
      g2l_b128(Bb + (size_t)row * C + kb + col, &sB[row * LROW + col]);
    }
    lds_fill_wait();

#pragma unroll
    for (int k2 = 0; k2 < KB; k2 += 32) {
      Frag fa[2], fb[4];
#pragma unroll
      for (int ms = 0; ms < 2; ++ms) {
        // 16-bit A 16x32 layout: lane half h, row l15: K = [h*8, h*8+8) and [16+h*8, ...)
        const bf16_t* p = &sA[(wm + ms * 16 + l15) * LROW + k2 + half * 8];
        fa[ms].u[0] = *(const uint4*)p;
        fa[ms].u[1] = *(const uint4*)(p + 16);
      }
#pragma unroll
      for (int ns = 0; ns < 4; ++ns) {
        // B 32x16 layout: lane half h, col l15: K = [h*8, h*8+8) and [16+h*8, ...)
        const bf16_t* p = &sB[(wn + ns * 16 + l15) * LROW + k2 + half * 8];
        fb[ns].u[0] = *(const uint4*)p;
        fb[ns].u[1] = *(const uint4*)(p + 16);
      }
#pragma unroll
      for (int ms = 0; ms < 2; ++ms)
#pragma unroll
        for (int ns = 0; ns < 4; ++ns)
          acc[ms][ns] = __builtin_amdgcn_wmma_f32_16x16x32_bf16(
              false, fa[ms].v, false, fb[ns].v, (short)0, acc[ms][ns], false, false);
    }
  }

  // Epilogue. C/D layout: VGPR r, lanes 0-15: (M=r, N=lane), lanes 16-31: (M=8+r, N=lane-16)
#pragma unroll
  for (int ns = 0; ns < 4; ++ns) {
    const int o = n0 + wn + ns * 16 + l15;
    float gg = 0.f, bb = 0.f;
    if (MODE == 0) bb = bias[o]; else gg = g[b * C + o];
#pragma unroll
    for (int ms = 0; ms < 2; ++ms) {
#pragma unroll
      for (int r = 0; r < 8; ++r) {
        const int hw = m0 + wm + ms * 16 + half * 8 + r;
        const size_t base = ((size_t)b * HW + hw) * C;
        float v = acc[ms][ns][r];
        if (MODE == 0) {
          bf16_t z = f2bf(v + bb);
          if (o < 128) { X1[base + 2 * o] = z; X1[base + 2 * o + 1] = z; }
          else { int k = o - 128; X2[base + 2 * k] = z; X2[base + 2 * k + 1] = z; }
        } else {
          float h  = gg * v;
          float ge = 0.5f * h * (1.0f + erff(h * 0.70710678118654752f));
          X2[base + o] = f2bf(bf2f(X2[base + o]) + ge);
        }
      }
    }
  }
}

// (b, c, hw) f32 -> (b, hw, c) bf16, LDS-tiled transpose
__global__ __launch_bounds__(256) void nchw_to_nhwc_bf16(
    const float* __restrict__ x, bf16_t* __restrict__ xT)
{
  __shared__ float tile[32][33];
  const int b = blockIdx.z, hw0 = blockIdx.x * 32, c0 = blockIdx.y * 32;
  const int tx = threadIdx.x & 31, ty = threadIdx.x >> 5;
#pragma unroll
  for (int j = 0; j < 4; ++j) {
    int c = c0 + ty + j * 8;
    tile[ty + j * 8][tx] = x[((size_t)b * C + c) * HW + hw0 + tx];
  }
  __syncthreads();
#pragma unroll
  for (int j = 0; j < 4; ++j) {
    int hw = hw0 + ty + j * 8;
    xT[((size_t)b * HW + hw) * C + c0 + tx] = f2bf(tile[tx][ty + j * 8]);
  }
}

__global__ __launch_bounds__(256) void cvt_bf16(
    const float* __restrict__ in, bf16_t* __restrict__ out, int n)
{
  int i = blockIdx.x * 256 + threadIdx.x;
  if (i < n) out[i] = f2bf(in[i]);
}

__global__ __launch_bounds__(256) void add_bf16(
    const bf16_t* __restrict__ a, const bf16_t* __restrict__ b,
    bf16_t* __restrict__ c, int n8)
{
  int i = blockIdx.x * 256 + threadIdx.x;
  if (i >= n8) return;
  v8bf x = ((const v8bf*)a)[i];
  v8bf y = ((const v8bf*)b)[i];
  v8bf r;
#pragma unroll
  for (int j = 0; j < 8; ++j) r[j] = f2bf(bf2f(x[j]) + bf2f(y[j]));
  ((v8bf*)c)[i] = r;
}

// g[b,o] = (s+1) * rsqrt((s+1)^2 * sum_c mod_w[o,c]^2 + eps), s = affine_w[o,:]·y[b,:] + affine_b[o]
__global__ __launch_bounds__(256) void mod_gain(
    const float* __restrict__ y, const float* __restrict__ aw,
    const float* __restrict__ ab, const float* __restrict__ mw,
    float* __restrict__ g)
{
  const int b = blockIdx.x, o = threadIdx.x;
  const float* w  = aw + (size_t)o * SD;
  const float* yb = y + (size_t)b * SD;
  float s = 0.f;
  for (int k = 0; k < SD; ++k) s = fmaf(w[k], yb[k], s);
  s += ab[o];
  const float* m = mw + (size_t)o * C;
  float q = 0.f;
  for (int k = 0; k < C; ++k) q = fmaf(m[k], m[k], q);
  float sp = s + 1.0f;
  g[b * C + o] = sp * rsqrtf(sp * sp * q + 1e-8f);
}

// out[b,c,2h+{0,1},2w+{0,1}] = 0.5*(x1[b,hw,c] + x2[b,hw,c])
__global__ __launch_bounds__(256) void final_upsample(
    const bf16_t* __restrict__ x1, const bf16_t* __restrict__ x2,
    float* __restrict__ out)
{
  int idx  = blockIdx.x * 256 + threadIdx.x;
  int hw   = idx & (HW - 1);
  int rest = idx >> 12;
  int c    = rest & (C - 1);
  int b    = rest >> 8;
  size_t sidx = ((size_t)b * HW + hw) * C + c;
  float v = 0.5f * (bf2f(x1[sidx]) + bf2f(x2[sidx]));
  int h = hw >> 6, w = hw & 63;
  float* o = out + (((size_t)(b * C + c)) * 128 + 2 * h) * 128 + 2 * w;
  float2 vv = make_float2(v, v);
  *(float2*)o = vv;
  *(float2*)(o + 128) = vv;
}

} // namespace

extern "C" void kernel_launch(void* const* d_in, const int* in_sizes, int n_in,
                              void* d_out, int out_size, void* d_ws, size_t ws_size,
                              hipStream_t stream) {
  (void)in_sizes; (void)n_in; (void)out_size; (void)ws_size;
  const float* x        = (const float*)d_in[0];
  const float* y        = (const float*)d_in[1];
  const float* ch_w     = (const float*)d_in[2];
  const float* ch_b     = (const float*)d_in[3];
  const float* affine_w = (const float*)d_in[4];
  const float* affine_b = (const float*)d_in[5];
  const float* mod_w    = (const float*)d_in[6];

  char* ws = (char*)d_ws;
  const size_t SZ = (size_t)NB * HW * C * sizeof(bf16_t); // 8 MiB per state tensor
  bf16_t* xT   = (bf16_t*)(ws);
  bf16_t* bufA = (bf16_t*)(ws + SZ);
  bf16_t* x2   = (bf16_t*)(ws + 2 * SZ);
  bf16_t* bufB = (bf16_t*)(ws + 3 * SZ);
  bf16_t* w1b  = (bf16_t*)(ws + 4 * SZ);
  bf16_t* mwb  = (bf16_t*)(ws + 4 * SZ + (size_t)C * C * sizeof(bf16_t));
  float*  gws  = (float*) (ws + 4 * SZ + (size_t)7 * C * C * sizeof(bf16_t));

  // Precision/layout prep
  nchw_to_nhwc_bf16<<<dim3(HW / 32, C / 32, NB), 256, 0, stream>>>(x, xT);
  cvt_bf16<<<(C * C + 255) / 256, 256, 0, stream>>>(ch_w, w1b, C * C);
  cvt_bf16<<<(6 * C * C + 255) / 256, 256, 0, stream>>>(mod_w, mwb, 6 * C * C);

  dim3 ggrid(HW / TM, C / TN, NB);
  // 1x1 conv (fused bias + channel-duplication scatter into x1/x2)
  gemm_wmma<0><<<ggrid, 256, 0, stream>>>(xT, w1b, ch_b, nullptr, bufA, x2);

  bf16_t* x1 = bufA;
  bf16_t* y1 = bufB;
  const int n8 = NB * HW * C / 8;
  for (int i = 0; i < 6; ++i) {
    add_bf16<<<(n8 + 255) / 256, 256, 0, stream>>>(x1, x2, y1, n8);
    mod_gain<<<NB, C, 0, stream>>>(y, affine_w + (size_t)i * C * SD,
                                   affine_b + (size_t)i * C,
                                   mod_w + (size_t)i * C * C, gws);
    gemm_wmma<1><<<ggrid, 256, 0, stream>>>(y1, mwb + (size_t)i * C * C,
                                            nullptr, gws, nullptr, x2);
    bf16_t* tmp = x1; x1 = y1; y1 = tmp;
  }
  final_upsample<<<(NB * HW * C) / 256, 256, 0, stream>>>(x1, x2, (float*)d_out);
}
